// ThreatTriageHead_68779606278943
// MI455X (gfx1250) — matrix operand order repaired and checked by reading, not compiled, not run on hardware
//
#include <hip/hip_runtime.h>
#include <cstdint>
#include <cstddef>

// ---------------------------------------------------------------------------
// ThreatTriageHead — fused WMMA implementation for gfx1250 (MI455X)
// B=4096, D=1024, H=256, H2=128, H3=64, C=3, S=100 MC samples
// Output: 4096 x 20 f32
// Stage2 is sample-chunked 4x (1024 WGs) for wave32/WGP occupancy; partial
// MC statistics are reduced by a finalize kernel.
// ---------------------------------------------------------------------------

typedef _Float16 h16;
typedef __attribute__((ext_vector_type(8)))  _Float16 v8h;
typedef __attribute__((ext_vector_type(16))) _Float16 v16h;
typedef __attribute__((ext_vector_type(8)))  float    v8f;

#define REF_EPS 1e-8f

// keep-prob thresholds: (uint32_t)(p * 2^32)
constexpr uint32_t T_K070 = 3006477107u;  // 0.70
constexpr uint32_t T_K079 = 3393024983u;  // 0.79
constexpr uint32_t T_K085 = 3650722201u;  // 0.85
constexpr uint32_t T_K080 = 3435973836u;  // 0.80
constexpr uint32_t SEED_E1 = 0x243F6A88u;
constexpr uint32_t SEED_E2 = 0x85A308D3u;
constexpr uint32_t SEED_M1 = 0x13198A2Eu;
constexpr uint32_t SEED_M2 = 0x03707344u;
constexpr uint32_t SEED_M3 = 0xA4093822u;

__device__ __forceinline__ uint32_t hashu(uint32_t x) {
  x ^= x >> 16; x *= 0x7feb352dU;
  x ^= x >> 15; x *= 0x846ca68bU;
  x ^= x >> 16; return x;
}

__device__ __forceinline__ v8f vzero8() {
  v8f z;
#pragma unroll
  for (int i = 0; i < 8; ++i) z[i] = 0.f;
  return z;
}

// A fragment: 16xK row-major tile. ISA layout: lanes 0-15 row M=lane,
// elems 0..7 -> K = kb + (lane>=16 ? 8:0) + j ; elems 8..15 -> +16.
__device__ __forceinline__ v16h ldfragA(const h16* base, int row, int stride,
                                        int kb, int lane) {
  const h16* p = base + row * stride + kb + ((lane >> 4) << 3);
  v8h lo = *(const v8h*)(p);
  v8h hi = *(const v8h*)(p + 16);
  v16h f;
#pragma unroll
  for (int i = 0; i < 8; ++i) { f[i] = lo[i]; f[i + 8] = hi[i]; }
  return f;
}

// B fragment from transposed (NxK) weights. Lane n = lane&15 fixed column,
// elems 0..7 -> K = kb + (lane>=16 ? 8:0) + j ; elems 8..15 -> +16.
__device__ __forceinline__ v16h ldfragB(const h16* baseT, int col, int stride,
                                        int kb, int lane) {
  const h16* p = baseT + col * stride + kb + ((lane >> 4) << 3);
  v8h lo = *(const v8h*)(p);
  v8h hi = *(const v8h*)(p + 16);
  v16h f;
#pragma unroll
  for (int i = 0; i < 8; ++i) { f[i] = lo[i]; f[i + 8] = hi[i]; }
  return f;
}

__device__ __forceinline__ v8f wmma_f16(v16h a, v16h b, v8f c) {
  return __builtin_amdgcn_wmma_f32_16x16x32_f16(false, a, false, b, (short)0,
                                                c, false, false);
}

// ---------------------------------------------------------------------------
// Pack kernels
// ---------------------------------------------------------------------------
__global__ void k_pack_transpose(const float* __restrict__ src,
                                 h16* __restrict__ dst, int K, int N) {
  int i = blockIdx.x * 256 + threadIdx.x;
  if (i >= K * N) return;
  int n = i / K, k = i - n * K;
  dst[n * K + k] = (h16)src[k * N + n];  // dst is [N][K]
}

__global__ void k_pack_w4(const float* __restrict__ W4, h16* __restrict__ dst) {
  int i = blockIdx.x * 256 + threadIdx.x;  // 16*64
  if (i >= 16 * 64) return;
  int n = i >> 6, k = i & 63;
  dst[i] = (n < 3) ? (h16)W4[k * 3 + n] : (h16)0.f;
}

__global__ void k_bn_fold(const float* __restrict__ g, const float* __restrict__ be,
                          const float* __restrict__ m, const float* __restrict__ v,
                          const float* __restrict__ b, float* __restrict__ scale,
                          float* __restrict__ shift, int n) {
  int i = blockIdx.x * 256 + threadIdx.x;
  if (i >= n) return;
  float sc = g[i] * rsqrtf(v[i] + 1e-5f);
  scale[i] = sc;
  shift[i] = (b[i] - m[i]) * sc + be[i];
}

// ---------------------------------------------------------------------------
// Stage 1: four first-layer GEMMs sharing the x tile.
// h1 = relu(bn(x@W1+b1)); e1d = drop0.7(relu(x@We1)); a1 = relu(x@Wa1);
// s1 = relu(x@Ws1).  16 rows/WG, 8 waves, 52 N-tiles.
// ---------------------------------------------------------------------------
__global__ __launch_bounds__(256) void k_stage1(
    const float* __restrict__ x,
    const h16* __restrict__ W1p, const h16* __restrict__ We1p,
    const h16* __restrict__ Wa1p, const h16* __restrict__ Ws1p,
    const float* __restrict__ scale1, const float* __restrict__ shift1,
    const float* __restrict__ eb1, const float* __restrict__ ab1,
    const float* __restrict__ sb1,
    h16* __restrict__ h1o, h16* __restrict__ e1o, h16* __restrict__ a1o,
    h16* __restrict__ s1o) {
  __shared__ __align__(16) h16 xs[16][1024];  // 32 KB
  const int t = threadIdx.x;
  const int row0 = blockIdx.x * 16;
  for (int i = t; i < 4096; i += 256) {  // 16*1024 in float4 chunks
    int r = i >> 8;
    int c = (i & 255) << 2;
    float4 v = *(const float4*)(x + (size_t)(row0 + r) * 1024 + c);
    xs[r][c + 0] = (h16)v.x; xs[r][c + 1] = (h16)v.y;
    xs[r][c + 2] = (h16)v.z; xs[r][c + 3] = (h16)v.w;
  }
  __syncthreads();
  const int lane = t & 31;
  const int wave = t >> 5;
  const int nr = lane & 15;
  for (int tile = wave; tile < 52; tile += 8) {
    const h16* Wp; int nt, kind;
    if (tile < 16)      { Wp = W1p;  nt = tile;      kind = 0; }
    else if (tile < 32) { Wp = We1p; nt = tile - 16; kind = 1; }
    else if (tile < 48) { Wp = Wa1p; nt = tile - 32; kind = 2; }
    else                { Wp = Ws1p; nt = tile - 48; kind = 3; }
    v8f acc = vzero8();
    for (int kb = 0; kb < 1024; kb += 32) {
      v16h a = ldfragA(&xs[0][0], nr, 1024, kb, lane);
      v16h b = ldfragB(Wp, nt * 16 + nr, 1024, kb, lane);
      acc = wmma_f16(a, b, acc);
    }
    const int n = nt * 16 + nr;
#pragma unroll
    for (int r = 0; r < 8; ++r) {
      const int m = r + ((lane >> 4) << 3);
      const int grow = row0 + m;
      float v = acc[r];
      if (kind == 0) {
        v = fmaxf(v * scale1[n] + shift1[n], 0.f);
        h1o[(size_t)grow * 256 + n] = (h16)v;
      } else if (kind == 1) {
        v = fmaxf(v + eb1[n], 0.f);
        uint32_t idx = (uint32_t)grow * 256u + (uint32_t)n;
        v = (hashu(SEED_E1 ^ idx) <= T_K070) ? v * (1.f / 0.7f) : 0.f;
        e1o[(size_t)grow * 256 + n] = (h16)v;
      } else if (kind == 2) {
        v = fmaxf(v + ab1[n], 0.f);
        a1o[(size_t)grow * 256 + n] = (h16)v;
      } else {
        v = fmaxf(v + sb1[n], 0.f);
        s1o[(size_t)grow * 64 + n] = (h16)v;
      }
    }
  }
}

// ---------------------------------------------------------------------------
// Stage 2: MC chain fully fused in LDS, sample-chunked across gridDim.y = 4.
// chunk 0: s = 0..25 (writes logits/probs of s==0), chunk c>0: 25 samples.
// Partial MC sums (Σp, Σp², Σent over s>=1) -> part[chunk][row][8].
// ---------------------------------------------------------------------------
__global__ __launch_bounds__(256) void k_stage2(
    const h16* __restrict__ h1b, const h16* __restrict__ W2p,
    const h16* __restrict__ W3p, const h16* __restrict__ W4p,
    const float* __restrict__ sc2g, const float* __restrict__ sh2g,
    const float* __restrict__ b3, const float* __restrict__ b4,
    float* __restrict__ part, float* __restrict__ out) {
  __shared__ __align__(16) h16 h1s[16][256];  // 8 KB
  __shared__ __align__(16) h16 d1s[16][256];  // 8 KB
  __shared__ __align__(16) h16 h2s[16][128];  // 4 KB
  __shared__ __align__(16) h16 h3s[16][64];   // 2 KB
  __shared__ float sc2[128], sh2[128];
  __shared__ float lg[16][4];

  const int t = threadIdx.x;
  const int lane = t & 31;
  const int wave = t >> 5;
  const int nr = lane & 15;
  const int row0 = blockIdx.x * 16;
  const int chunk = blockIdx.y;                   // 0..3
  const int sBeg = (chunk == 0) ? 0 : (1 + 25 * chunk);
  const int sEnd = 1 + 25 * (chunk + 1);          // 26 / 51 / 76 / 101

  for (int i = t; i < 16 * 256 / 8; i += 256)
    ((v8h*)&h1s[0][0])[i] = ((const v8h*)(h1b + (size_t)row0 * 256))[i];
  if (t < 128) { sc2[t] = sc2g[t]; sh2[t] = sh2g[t]; }
  __syncthreads();

  float sP[3] = {0.f, 0.f, 0.f};
  float sQ[3] = {0.f, 0.f, 0.f};
  float sEnt = 0.f;

  for (int s = sBeg; s < sEnd; ++s) {
    // d1 = dropout(h1, keep=0.7), mask indexed by absolute sample id
    for (int i = t; i < 4096; i += 256) {
      int r = i >> 8, c = i & 255;
      float hv = (float)h1s[r][c];
      uint32_t idx = (uint32_t)s * 1048576u + (uint32_t)(row0 + r) * 256u + (uint32_t)c;
      hv = (hashu(SEED_M1 ^ idx) <= T_K070) ? hv * (1.f / 0.7f) : 0.f;
      d1s[r][c] = (h16)hv;
    }
    __syncthreads();
    // layer 2: 16x256 @ 256x128 ; 8 waves, 1 N-tile each
    {
      v8f acc = vzero8();
#pragma unroll
      for (int kb = 0; kb < 256; kb += 32) {
        v16h a = ldfragA(&d1s[0][0], nr, 256, kb, lane);
        v16h b = ldfragB(W2p, wave * 16 + nr, 256, kb, lane);
        acc = wmma_f16(a, b, acc);
      }
      const int n = wave * 16 + nr;
      const float sc = sc2[n], sh = sh2[n];
#pragma unroll
      for (int r = 0; r < 8; ++r) {
        int m = r + ((lane >> 4) << 3);
        float v = fmaxf(acc[r] * sc + sh, 0.f);
        uint32_t idx = (uint32_t)s * 524288u + (uint32_t)(row0 + m) * 128u + (uint32_t)n;
        v = (hashu(SEED_M2 ^ idx) <= T_K079) ? v * (1.f / 0.79f) : 0.f;
        h2s[m][n] = (h16)v;
      }
    }
    __syncthreads();
    // layer 3: 16x128 @ 128x64 ; waves 0-3
    if (wave < 4) {
      v8f acc = vzero8();
#pragma unroll
      for (int kb = 0; kb < 128; kb += 32) {
        v16h a = ldfragA(&h2s[0][0], nr, 128, kb, lane);
        v16h b = ldfragB(W3p, wave * 16 + nr, 128, kb, lane);
        acc = wmma_f16(a, b, acc);
      }
      const int n = wave * 16 + nr;
      const float bb = b3[n];
#pragma unroll
      for (int r = 0; r < 8; ++r) {
        int m = r + ((lane >> 4) << 3);
        float v = fmaxf(acc[r] + bb, 0.f);
        uint32_t idx = (uint32_t)s * 262144u + (uint32_t)(row0 + m) * 64u + (uint32_t)n;
        v = (hashu(SEED_M3 ^ idx) <= T_K085) ? v * (1.f / 0.85f) : 0.f;
        h3s[m][n] = (h16)v;
      }
    }
    __syncthreads();
    // layer 4: 16x64 @ 64x16 (3 real cols) ; wave 0 only
    if (wave == 0) {
      v8f acc = vzero8();
#pragma unroll
      for (int kb = 0; kb < 64; kb += 32) {
        v16h a = ldfragA(&h3s[0][0], nr, 64, kb, lane);
        v16h b = ldfragB(W4p, nr, 64, kb, lane);
        acc = wmma_f16(a, b, acc);
      }
      if (nr < 3) {
        const float bb = b4[nr];
#pragma unroll
        for (int r = 0; r < 8; ++r) {
          int m = r + ((lane >> 4) << 3);
          lg[m][nr] = acc[r] + bb;
        }
      }
    }
    __syncthreads();
    // softmax + MC stats (one thread per row)
    if (t < 16) {
      float l0 = lg[t][0], l1 = lg[t][1], l2 = lg[t][2];
      float mx = fmaxf(l0, fmaxf(l1, l2));
      float e0 = expf(l0 - mx), e1 = expf(l1 - mx), e2 = expf(l2 - mx);
      float inv = 1.f / (e0 + e1 + e2);
      float p0 = e0 * inv, p1 = e1 * inv, p2 = e2 * inv;
      if (s == 0) {
        float* o = out + (size_t)(row0 + t) * 20;
        o[0] = l0; o[1] = l1; o[2] = l2;
        o[3] = p0; o[4] = p1; o[5] = p2;
      } else {
        sP[0] += p0; sP[1] += p1; sP[2] += p2;
        sQ[0] += p0 * p0; sQ[1] += p1 * p1; sQ[2] += p2 * p2;
        sEnt -= p0 * logf(p0 + REF_EPS) + p1 * logf(p1 + REF_EPS) +
                p2 * logf(p2 + REF_EPS);
      }
    }
  }
  if (t < 16) {
    float* pp = part + ((size_t)chunk * 4096 + (row0 + t)) * 8;
    pp[0] = sP[0]; pp[1] = sP[1]; pp[2] = sP[2];
    pp[3] = sQ[0]; pp[4] = sQ[1]; pp[5] = sQ[2];
    pp[6] = sEnt;  pp[7] = 0.f;
  }
}

// Reduce the 4 sample-chunk partials -> cols 10..17.
__global__ __launch_bounds__(256) void k_stage2fin(
    const float* __restrict__ part, float* __restrict__ out) {
  int row = blockIdx.x * 256 + threadIdx.x;
  if (row >= 4096) return;
  float sP[3] = {0.f, 0.f, 0.f}, sQ[3] = {0.f, 0.f, 0.f}, sEnt = 0.f;
#pragma unroll
  for (int c = 0; c < 4; ++c) {
    const float* pp = part + ((size_t)c * 4096 + row) * 8;
    sP[0] += pp[0]; sP[1] += pp[1]; sP[2] += pp[2];
    sQ[0] += pp[3]; sQ[1] += pp[4]; sQ[2] += pp[5];
    sEnt += pp[6];
  }
  float mean[3], stdv[3];
#pragma unroll
  for (int c = 0; c < 3; ++c) {
    mean[c] = sP[c] * 0.01f;
    float var = (sQ[c] - 100.f * mean[c] * mean[c]) * (1.f / 99.f);
    stdv[c] = sqrtf(fmaxf(var, 0.f));
  }
  float ment = -(mean[0] * logf(mean[0] + REF_EPS) +
                 mean[1] * logf(mean[1] + REF_EPS) +
                 mean[2] * logf(mean[2] + REF_EPS));
  float mi = ment - sEnt * 0.01f;
  float* o = out + (size_t)row * 20;
  o[10] = mean[0]; o[11] = mean[1]; o[12] = mean[2];
  o[13] = stdv[0]; o[14] = stdv[1]; o[15] = stdv[2];
  o[16] = ment;    o[17] = mi;
}

// ---------------------------------------------------------------------------
// Stage 3: epistemic / aleatoric / severity branches.
// Writes cols 6 (sev), 7 (epi), 8 (ale), 9 (total_unc).
// ---------------------------------------------------------------------------
__global__ __launch_bounds__(256) void k_stage3(
    const h16* __restrict__ e1b, const h16* __restrict__ a1b,
    const h16* __restrict__ s1b,
    const h16* __restrict__ We2p, const h16* __restrict__ Wa2p,
    const h16* __restrict__ Ws2p,
    const float* __restrict__ eb2, const float* __restrict__ ab2,
    const float* __restrict__ sb2,
    const float* __restrict__ We3, const float* __restrict__ eb3,
    const float* __restrict__ Wa3, const float* __restrict__ ab3,
    const float* __restrict__ Ws3, const float* __restrict__ sb3,
    float* __restrict__ out) {
  __shared__ __align__(16) h16 e1s[16][256];  // 8 KB
  __shared__ __align__(16) h16 a1s[16][256];  // 8 KB
  __shared__ __align__(16) h16 s1s[16][64];   // 2 KB
  __shared__ __align__(16) h16 e2s[16][128];  // 4 KB
  __shared__ __align__(16) h16 a2s[16][128];  // 4 KB
  __shared__ __align__(16) h16 s2s[16][32];   // 1 KB

  const int t = threadIdx.x;
  const int lane = t & 31;
  const int wave = t >> 5;
  const int nr = lane & 15;
  const int row0 = blockIdx.x * 16;

  for (int i = t; i < 512; i += 256)
    ((v8h*)&e1s[0][0])[i] = ((const v8h*)(e1b + (size_t)row0 * 256))[i];
  for (int i = t; i < 512; i += 256)
    ((v8h*)&a1s[0][0])[i] = ((const v8h*)(a1b + (size_t)row0 * 256))[i];
  for (int i = t; i < 128; i += 256)
    ((v8h*)&s1s[0][0])[i] = ((const v8h*)(s1b + (size_t)row0 * 64))[i];
  __syncthreads();

  for (int tile = wave; tile < 18; tile += 8) {
    v8f acc = vzero8();
    if (tile < 8) {           // e2 = drop0.8(relu(e1d @ We2 + eb2))
      for (int kb = 0; kb < 256; kb += 32) {
        v16h a = ldfragA(&e1s[0][0], nr, 256, kb, lane);
        v16h b = ldfragB(We2p, tile * 16 + nr, 256, kb, lane);
        acc = wmma_f16(a, b, acc);
      }
      const int n = tile * 16 + nr;
      const float bb = eb2[n];
#pragma unroll
      for (int r = 0; r < 8; ++r) {
        int m = r + ((lane >> 4) << 3);
        float v = fmaxf(acc[r] + bb, 0.f);
        uint32_t idx = (uint32_t)(row0 + m) * 128u + (uint32_t)n;
        v = (hashu(SEED_E2 ^ idx) <= T_K080) ? v * (1.f / 0.8f) : 0.f;
        e2s[m][n] = (h16)v;
      }
    } else if (tile < 16) {   // a2 = relu(a1 @ Wa2 + ab2)
      const int tt = tile - 8;
      for (int kb = 0; kb < 256; kb += 32) {
        v16h a = ldfragA(&a1s[0][0], nr, 256, kb, lane);
        v16h b = ldfragB(Wa2p, tt * 16 + nr, 256, kb, lane);
        acc = wmma_f16(a, b, acc);
      }
      const int n = tt * 16 + nr;
      const float bb = ab2[n];
#pragma unroll
      for (int r = 0; r < 8; ++r) {
        int m = r + ((lane >> 4) << 3);
        a2s[m][n] = (h16)fmaxf(acc[r] + bb, 0.f);
      }
    } else {                  // s2 = relu(s1 @ Ws2 + sb2), K=64, N=32
      const int tt = tile - 16;
      for (int kb = 0; kb < 64; kb += 32) {
        v16h a = ldfragA(&s1s[0][0], nr, 64, kb, lane);
        v16h b = ldfragB(Ws2p, tt * 16 + nr, 64, kb, lane);
        acc = wmma_f16(a, b, acc);
      }
      const int n = tt * 16 + nr;
      const float bb = sb2[n];
#pragma unroll
      for (int r = 0; r < 8; ++r) {
        int m = r + ((lane >> 4) << 3);
        s2s[m][n] = (h16)fmaxf(acc[r] + bb, 0.f);
      }
    }
  }
  __syncthreads();
  if (t < 16) {
    float se = eb3[0], sa = ab3[0], ss = sb3[0];
    for (int k = 0; k < 128; ++k) se += (float)e2s[t][k] * We3[k];
    for (int k = 0; k < 128; ++k) sa += (float)a2s[t][k] * Wa3[k];
    for (int k = 0; k < 32; ++k)  ss += (float)s2s[t][k] * Ws3[k];
    float epi = 1.f / (1.f + expf(-se));
    float ale = (sa > 20.f) ? sa : log1pf(expf(sa));
    float sev = 1.f / (1.f + expf(-ss));
    float tot = sqrtf(epi * epi + ale * ale);
    float* o = out + (size_t)(row0 + t) * 20;
    o[6] = sev; o[7] = epi; o[8] = ale; o[9] = tot;
  }
}

// ---------------------------------------------------------------------------
// Stage 4: calibration MLP (6->32->1) + risk. 1 thread / row.
// Writes cols 18 (calibrated), 19 (risk). Reads cols written by stages 2/3.
// ---------------------------------------------------------------------------
__global__ __launch_bounds__(256) void k_stage4(
    const float* __restrict__ Wc1, const float* __restrict__ cb1,
    const float* __restrict__ Wc2, const float* __restrict__ cb2,
    float* __restrict__ out) {
  int row = blockIdx.x * 256 + threadIdx.x;
  if (row >= 4096) return;
  float* o = out + (size_t)row * 20;
  float in6[6] = {o[0], o[1], o[2], o[7], o[8], o[16]};
  float c2 = cb2[0];
  for (int j = 0; j < 32; ++j) {
    float h = cb1[j];
#pragma unroll
    for (int i = 0; i < 6; ++i) h += in6[i] * Wc1[i * 32 + j];
    c2 += fmaxf(h, 0.f) * Wc2[j];
  }
  float calib = 1.f / (1.f + expf(-c2));
  float risk = (o[3] * 0.1f + o[4] * 0.5f + o[5] * 1.0f) * o[6];
  o[18] = calib;
  o[19] = risk;
}

// ---------------------------------------------------------------------------
// Host launcher
// ---------------------------------------------------------------------------
extern "C" void kernel_launch(void* const* d_in, const int* in_sizes, int n_in,
                              void* d_out, int out_size, void* d_ws,
                              size_t ws_size, hipStream_t stream) {
  const float* x   = (const float*)d_in[0];
  const float* W1  = (const float*)d_in[1];
  const float* b1  = (const float*)d_in[2];
  const float* g1  = (const float*)d_in[3];
  const float* be1 = (const float*)d_in[4];
  const float* m1  = (const float*)d_in[5];
  const float* v1  = (const float*)d_in[6];
  const float* W2  = (const float*)d_in[7];
  const float* b2  = (const float*)d_in[8];
  const float* g2  = (const float*)d_in[9];
  const float* be2 = (const float*)d_in[10];
  const float* m2  = (const float*)d_in[11];
  const float* v2  = (const float*)d_in[12];
  const float* W3  = (const float*)d_in[13];
  const float* b3  = (const float*)d_in[14];
  const float* W4  = (const float*)d_in[15];
  const float* b4  = (const float*)d_in[16];
  const float* We1 = (const float*)d_in[17];
  const float* eb1 = (const float*)d_in[18];
  const float* We2 = (const float*)d_in[19];
  const float* eb2 = (const float*)d_in[20];
  const float* We3 = (const float*)d_in[21];
  const float* eb3 = (const float*)d_in[22];
  const float* Wa1 = (const float*)d_in[23];
  const float* ab1 = (const float*)d_in[24];
  const float* Wa2 = (const float*)d_in[25];
  const float* ab2 = (const float*)d_in[26];
  const float* Wa3 = (const float*)d_in[27];
  const float* ab3 = (const float*)d_in[28];
  const float* Ws1 = (const float*)d_in[29];
  const float* sb1 = (const float*)d_in[30];
  const float* Ws2 = (const float*)d_in[31];
  const float* sb2 = (const float*)d_in[32];
  const float* Ws3 = (const float*)d_in[33];
  const float* sb3 = (const float*)d_in[34];
  const float* Wc1 = (const float*)d_in[35];
  const float* cb1 = (const float*)d_in[36];
  const float* Wc2 = (const float*)d_in[37];
  const float* cb2 = (const float*)d_in[38];
  float* out = (float*)d_out;

  uintptr_t base = (uintptr_t)d_ws;
  size_t off = 0;
  auto alloc = [&](size_t bytes) -> void* {
    void* p = (void*)(base + off);
    off = (off + bytes + 255) & ~(size_t)255;
    return p;
  };
  h16* W1p  = (h16*)alloc(256 * 1024 * sizeof(h16));
  h16* We1p = (h16*)alloc(256 * 1024 * sizeof(h16));
  h16* Wa1p = (h16*)alloc(256 * 1024 * sizeof(h16));
  h16* Ws1p = (h16*)alloc(64 * 1024 * sizeof(h16));
  h16* W2p  = (h16*)alloc(128 * 256 * sizeof(h16));
  h16* We2p = (h16*)alloc(128 * 256 * sizeof(h16));
  h16* Wa2p = (h16*)alloc(128 * 256 * sizeof(h16));
  h16* W3p  = (h16*)alloc(64 * 128 * sizeof(h16));
  h16* Ws2p = (h16*)alloc(32 * 64 * sizeof(h16));
  h16* W4p  = (h16*)alloc(16 * 64 * sizeof(h16));
  float* scale1 = (float*)alloc(256 * sizeof(float));
  float* shift1 = (float*)alloc(256 * sizeof(float));
  float* scale2 = (float*)alloc(128 * sizeof(float));
  float* shift2 = (float*)alloc(128 * sizeof(float));
  h16* h1b = (h16*)alloc((size_t)4096 * 256 * sizeof(h16));
  h16* e1b = (h16*)alloc((size_t)4096 * 256 * sizeof(h16));
  h16* a1b = (h16*)alloc((size_t)4096 * 256 * sizeof(h16));
  h16* s1b = (h16*)alloc((size_t)4096 * 64 * sizeof(h16));
  float* part = (float*)alloc((size_t)4 * 4096 * 8 * sizeof(float));

  auto packT = [&](const float* src, h16* dst, int K, int N) {
    int n = K * N;
    k_pack_transpose<<<(n + 255) / 256, 256, 0, stream>>>(src, dst, K, N);
  };
  packT(W1, W1p, 1024, 256);
  packT(We1, We1p, 1024, 256);
  packT(Wa1, Wa1p, 1024, 256);
  packT(Ws1, Ws1p, 1024, 64);
  packT(W2, W2p, 256, 128);
  packT(We2, We2p, 256, 128);
  packT(Wa2, Wa2p, 256, 128);
  packT(W3, W3p, 128, 64);
  packT(Ws2, Ws2p, 64, 32);
  k_pack_w4<<<4, 256, 0, stream>>>(W4, W4p);
  k_bn_fold<<<1, 256, 0, stream>>>(g1, be1, m1, v1, b1, scale1, shift1, 256);
  k_bn_fold<<<1, 256, 0, stream>>>(g2, be2, m2, v2, b2, scale2, shift2, 128);

  k_stage1<<<256, 256, 0, stream>>>(x, W1p, We1p, Wa1p, Ws1p, scale1, shift1,
                                    eb1, ab1, sb1, h1b, e1b, a1b, s1b);
  k_stage2<<<dim3(256, 4), 256, 0, stream>>>(h1b, W2p, W3p, W4p, scale2,
                                             shift2, b3, b4, part, out);
  k_stage2fin<<<16, 256, 0, stream>>>(part, out);
  k_stage3<<<256, 256, 0, stream>>>(e1b, a1b, s1b, We2p, Wa2p, Ws2p, eb2, ab2,
                                    sb2, We3, eb3, Wa3, ab3, Ws3, sb3, out);
  k_stage4<<<16, 256, 0, stream>>>(Wc1, cb1, Wc2, cb2, out);
}
